// SetAbstraction_74852690035264
// MI455X (gfx1250) — matrix operand order repaired and checked by reading, not compiled, not run on hardware
//
#include <hip/hip_runtime.h>
#include <math.h>

// ---------------- problem constants (match reference) ----------------
static constexpr int BB   = 8;
static constexpr int NPTS = 8192;
static constexpr int MC   = 2048;   // centroids
static constexpr int KNB  = 32;     // neighbors
static constexpr int CIN  = 64;
static constexpr int COUT = 128;
static constexpr int PE   = 16;
static constexpr int CG   = CIN + 3 + PE;   // 83
static constexpr int CGP  = 96;             // padded to 3 k-tiles of 32
static constexpr int HID  = 64;
static constexpr float R2 = 0.2f * 0.2f;

typedef __attribute__((ext_vector_type(16))) __bf16 v16bf;
typedef __attribute__((ext_vector_type(8)))  __bf16 v8bf;
typedef __attribute__((ext_vector_type(8)))  float  v8f;

// LDS byte offset from a generic pointer to a __shared__ object:
// flat LDS addresses are {aperture_hi32, lds_offset_lo32} (ISA 10.2).
__device__ inline unsigned lds_off(const void* p) {
  return (unsigned)(size_t)p;
}

// ---------------- WMMA fragment helpers (wave32, bf16 16x16x32) ------
// A 16x32 (MxK): lane<16 -> M=lane, K in {kb..kb+7, kb+16..kb+23}, kb=8*(lane>=16)
__device__ inline v16bf load_a_frag(const __bf16* base, int ld, int row0, int col0) {
  int lane = threadIdx.x & 31;
  int mr = lane & 15, half = lane >> 4;
  const __bf16* rp = base + (row0 + mr) * ld + col0 + half * 8;
  v8bf lo = *(const v8bf*)rp;          // K = kb+0..7   (16B aligned)
  v8bf hi = *(const v8bf*)(rp + 16);   // K = kb+16..23
  return __builtin_shufflevector(lo, hi, 0, 1, 2, 3, 4, 5, 6, 7,
                                 8, 9, 10, 11, 12, 13, 14, 15);
}

// B tiles pre-swizzled in LDS: tile*32 lanes * 16 contiguous bf16 each
__device__ inline v16bf load_b_tile(const __bf16* tiles, int tile) {
  int lane = threadIdx.x & 31;
  return *(const v16bf*)(tiles + ((tile * 32 + lane) << 4));
}

__device__ inline v8f wmma_bf16(v16bf a, v16bf b, v8f c) {
  return __builtin_amdgcn_wmma_f32_16x16x32_bf16(false, a, false, b, (short)0, c,
                                                 false, false);
}

__device__ inline float gelu_exact(float x) {
  return 0.5f * x * (1.0f + erff(x * 0.70710678118654752f));
}

// ---------------- kernel 1: farthest point sampling ------------------
__global__ void fps_kernel(const float* __restrict__ pos,
                           int* __restrict__ sidx,
                           float* __restrict__ newpos) {
  __shared__ float mind[NPTS];     // 32 KB
  __shared__ float rv[1024];
  __shared__ int   ri[1024];
  __shared__ float cp[3];

  const int b = blockIdx.x;
  const int t = threadIdx.x;
  const float* P = pos + (size_t)b * NPTS * 3;

  if (t == 0) { cp[0] = P[0]; cp[1] = P[1]; cp[2] = P[2]; }
  __syncthreads();

  for (int p = t; p < NPTS; p += 1024) {
    float dx = P[p * 3 + 0] - cp[0];
    float dy = P[p * 3 + 1] - cp[1];
    float dz = P[p * 3 + 2] - cp[2];
    mind[p] = dx * dx + dy * dy + dz * dz;
  }
  if (t == 0) {
    sidx[(size_t)b * MC] = 0;
    newpos[(size_t)b * MC * 3 + 0] = cp[0];
    newpos[(size_t)b * MC * 3 + 1] = cp[1];
    newpos[(size_t)b * MC * 3 + 2] = cp[2];
  }
  __syncthreads();

  for (int s = 1; s < MC; ++s) {
    float bv = -1.0f; int bi = NPTS;
    for (int p = t; p < NPTS; p += 1024) {
      float v = mind[p];
      if (v > bv) { bv = v; bi = p; }
    }
    rv[t] = bv; ri[t] = bi;
    __syncthreads();
#pragma unroll
    for (int off = 512; off > 0; off >>= 1) {
      if (t < off) {
        float ov = rv[t + off]; int oi = ri[t + off];
        if (ov > rv[t] || (ov == rv[t] && oi < ri[t])) { rv[t] = ov; ri[t] = oi; }
      }
      __syncthreads();
    }
    if (t == 0) {
      int i = ri[0];
      sidx[(size_t)b * MC + s] = i;
      cp[0] = P[i * 3 + 0]; cp[1] = P[i * 3 + 1]; cp[2] = P[i * 3 + 2];
      newpos[((size_t)b * MC + s) * 3 + 0] = cp[0];
      newpos[((size_t)b * MC + s) * 3 + 1] = cp[1];
      newpos[((size_t)b * MC + s) * 3 + 2] = cp[2];
    }
    __syncthreads();
    float cx = cp[0], cy = cp[1], cz = cp[2];
    for (int p = t; p < NPTS; p += 1024) {
      float dx = P[p * 3 + 0] - cx;
      float dy = P[p * 3 + 1] - cy;
      float dz = P[p * 3 + 2] - cz;
      float d = dx * dx + dy * dy + dz * dz;
      mind[p] = fminf(mind[p], d);
    }
    __syncthreads();
  }
}

// ---------------- kernel 2: ball query (one wave per centroid) -------
__global__ void ballq_kernel(const float* __restrict__ pos,
                             const float* __restrict__ newpos,
                             int* __restrict__ nidx) {
  const int wid = (blockIdx.x * blockDim.x + threadIdx.x) >> 5;
  const int lane = threadIdx.x & 31;
  if (wid >= BB * MC) return;
  const int b = wid >> 11;
  const float* P = pos + (size_t)b * NPTS * 3;
  const float cx = newpos[(size_t)wid * 3 + 0];
  const float cy = newpos[(size_t)wid * 3 + 1];
  const float cz = newpos[(size_t)wid * 3 + 2];

  int cnt = 0;
  int first = 0;
  const unsigned ltmask = (1u << lane) - 1u;
  for (int base = 0; base < NPTS && cnt < KNB; base += 32) {
    const int p = base + lane;
    float dx = P[p * 3 + 0] - cx;
    float dy = P[p * 3 + 1] - cy;
    float dz = P[p * 3 + 2] - cz;
    bool valid = (dx * dx + dy * dy + dz * dz) <= R2;
    unsigned mask = __builtin_amdgcn_ballot_w32(valid);
    if (cnt == 0 && mask) first = base + __ffs((int)mask) - 1;
    int slot = cnt + __popc(mask & ltmask);
    if (valid && slot < KNB) nidx[(size_t)wid * KNB + slot] = p;
    cnt += __popc(mask);
  }
  if (cnt < KNB) {
    for (int s = cnt + lane; s < KNB; s += 32) nidx[(size_t)wid * KNB + s] = first;
  }
}

// ---------------- kernel 3: grouped MLP via WMMA (1 wave / centroid) -
// X column order (permuted vs reference, compensated in W1 swizzle):
//   cols 0..63 = grouped features, 64..66 = rel xyz, 67..82 = pe, 83..95 = pad
__device__ inline int w1row_of_col(int k) {
  if (k < 64) return 3 + k;    // feature -> W1 rows 3..66
  if (k < 67) return k - 64;   // rel     -> W1 rows 0..2
  if (k < 83) return k;        // pe      -> W1 rows 67..82
  return -1;                   // pad
}

__global__ void mlp_kernel(const float* __restrict__ pos,
                           const float* __restrict__ feat,
                           const float* __restrict__ W1, const float* __restrict__ b1,
                           const float* __restrict__ g1, const float* __restrict__ be1,
                           const float* __restrict__ W2, const float* __restrict__ b2,
                           const float* __restrict__ g2, const float* __restrict__ be2,
                           const float* __restrict__ Wpe, const float* __restrict__ bpe,
                           const float* __restrict__ Wres, const float* __restrict__ bres,
                           const float* __restrict__ gres, const float* __restrict__ beres,
                           const int* __restrict__ sidx, const int* __restrict__ nidx,
                           const float* __restrict__ newpos,
                           float* __restrict__ outfeat) {
  __shared__ __bf16 sW1t[12 * 32 * 16];   // 12 KB  B-tiles, fragment order
  __shared__ __bf16 sW2t[16 * 32 * 16];   // 16 KB
  __shared__ __bf16 sX[KNB * CGP];        //  6 KB  (A staging; ld=64 in gemm2)
  __shared__ float  sH[KNB * HID];        //  8 KB  (async landing pad + gemm1 out)
  __shared__ float  sY[KNB * COUT];       // 16 KB
  __shared__ float  sB1[HID], sG1[HID], sBe1[HID];
  __shared__ float  sB2[COUT], sG2[COUT], sBe2[COUT];
  __shared__ float  sWpe[3 * PE], sBpe[PE];

  const int lane = threadIdx.x;           // blockDim == 32 (one wave)
  const int mr = lane & 15, half = lane >> 4;

  // ---- one-time staging: LN params, PE weights, swizzled B tiles ----
  for (int i = lane; i < HID; i += 32) { sB1[i] = b1[i]; sG1[i] = g1[i]; sBe1[i] = be1[i]; }
  for (int i = lane; i < COUT; i += 32) { sB2[i] = b2[i]; sG2[i] = g2[i]; sBe2[i] = be2[i]; }
  for (int i = lane; i < 3 * PE; i += 32) sWpe[i] = Wpe[i];
  for (int i = lane; i < PE; i += 32) sBpe[i] = bpe[i];

#pragma unroll
  for (int kt = 0; kt < 3; ++kt)
#pragma unroll
    for (int nt = 0; nt < 4; ++nt) {
      __bf16* dst = sW1t + (((kt * 4 + nt) * 32 + lane) << 4);
      const int colN = nt * 16 + mr;
      const int kb = kt * 32 + half * 8;
#pragma unroll
      for (int e = 0; e < 16; ++e) {
        int r = w1row_of_col(kb + e + (e & 8));
        dst[e] = (__bf16)(r >= 0 ? W1[r * HID + colN] : 0.0f);
      }
    }
#pragma unroll
  for (int kt = 0; kt < 2; ++kt)
#pragma unroll
    for (int nt = 0; nt < 8; ++nt) {
      __bf16* dst = sW2t + (((kt * 8 + nt) * 32 + lane) << 4);
      const int colN = nt * 16 + mr;
      const int kb = kt * 32 + half * 8;
#pragma unroll
      for (int e = 0; e < 16; ++e) dst[e] = (__bf16)W2[(kb + e + (e & 8)) * COUT + colN];
    }
  __syncthreads();

  for (int bm = blockIdx.x; bm < BB * MC; bm += gridDim.x) {
    const int b = bm >> 11;

    // ---- Phase A: async-gather feature row into sH; rel/pe tail ----
    {
      const int n = nidx[(size_t)bm * KNB + lane];
      const float* F = feat + ((size_t)b * NPTS + n) * CIN;   // 256 B
      unsigned loff = lds_off(sH + lane * HID);
      // 16 x b128 async global->LDS copies (ASYNCcnt)
#define SA_ALOAD(OFF)                                                          \
  asm volatile("global_load_async_to_lds_b128 %0, %1, off offset:" #OFF        \
               ::"v"(loff), "v"(F) : "memory")
      SA_ALOAD(0);   SA_ALOAD(16);  SA_ALOAD(32);  SA_ALOAD(48);
      SA_ALOAD(64);  SA_ALOAD(80);  SA_ALOAD(96);  SA_ALOAD(112);
      SA_ALOAD(128); SA_ALOAD(144); SA_ALOAD(160); SA_ALOAD(176);
      SA_ALOAD(192); SA_ALOAD(208); SA_ALOAD(224); SA_ALOAD(240);
#undef SA_ALOAD

      // overlap: rel xyz + positional encoding + pad -> sX cols 64..95
      const float* P = pos + ((size_t)b * NPTS + n) * 3;
      const float cx = newpos[(size_t)bm * 3 + 0];
      const float cy = newpos[(size_t)bm * 3 + 1];
      const float cz = newpos[(size_t)bm * 3 + 2];
      const float rx = P[0] - cx, ry = P[1] - cy, rz = P[2] - cz;
      float tv[32];
      tv[0] = rx; tv[1] = ry; tv[2] = rz;
#pragma unroll
      for (int j = 0; j < PE; ++j)
        tv[3 + j] = rx * sWpe[j] + ry * sWpe[PE + j] + rz * sWpe[2 * PE + j] + sBpe[j];
#pragma unroll
      for (int c = 19; c < 32; ++c) tv[c] = 0.0f;
      __bf16* xr = sX + lane * CGP;
      v8bf* tdst = (v8bf*)(xr + 64);
#pragma unroll
      for (int i = 0; i < 4; ++i) {
        v8bf t;
#pragma unroll
        for (int j = 0; j < 8; ++j) t[j] = (__bf16)tv[8 * i + j];
        tdst[i] = t;
      }

      // wait for async engine, then convert f32 row -> bf16 cols 0..63
      asm volatile("s_wait_asynccnt 0" ::: "memory");
      __syncthreads();
      const float* srow = sH + lane * HID;
      v8bf* fdst = (v8bf*)xr;
#pragma unroll
      for (int i = 0; i < 8; ++i) {
        float4 q0 = *(const float4*)(srow + 8 * i);
        float4 q1 = *(const float4*)(srow + 8 * i + 4);
        v8bf t;
        t[0] = (__bf16)q0.x; t[1] = (__bf16)q0.y; t[2] = (__bf16)q0.z; t[3] = (__bf16)q0.w;
        t[4] = (__bf16)q1.x; t[5] = (__bf16)q1.y; t[6] = (__bf16)q1.z; t[7] = (__bf16)q1.w;
        fdst[i] = t;
      }
    }
    __syncthreads();

    // ---- Phase B: GEMM1  X[32x96] @ W1[96x64] ----
    {
      v8f acc[2][4];
#pragma unroll
      for (int i = 0; i < 2; ++i)
#pragma unroll
        for (int j = 0; j < 4; ++j)
          acc[i][j] = (v8f){0.f, 0.f, 0.f, 0.f, 0.f, 0.f, 0.f, 0.f};
#pragma unroll
      for (int kt = 0; kt < 3; ++kt) {
        v16bf a0 = load_a_frag(sX, CGP, 0, kt * 32);
        v16bf a1 = load_a_frag(sX, CGP, 16, kt * 32);
#pragma unroll
        for (int nt = 0; nt < 4; ++nt) {
          v16bf bf = load_b_tile(sW1t, kt * 4 + nt);
          acc[0][nt] = wmma_bf16(a0, bf, acc[0][nt]);
          acc[1][nt] = wmma_bf16(a1, bf, acc[1][nt]);
        }
      }
#pragma unroll
      for (int mt = 0; mt < 2; ++mt)
#pragma unroll
        for (int nt = 0; nt < 4; ++nt)
#pragma unroll
          for (int r = 0; r < 8; ++r)
            sH[(mt * 16 + r + half * 8) * HID + nt * 16 + mr] = acc[mt][nt][r];
    }
    __syncthreads();

    // ---- Phase C: +b1, LayerNorm, GELU (two-pass, register-light) ----
    {
      const float* hr = sH + lane * HID;
      float sum = 0.f, sq = 0.f;
#pragma unroll
      for (int i = 0; i < HID / 4; ++i) {
        float4 q = *(const float4*)(hr + 4 * i);
        float x0 = q.x + sB1[4 * i + 0];
        float x1 = q.y + sB1[4 * i + 1];
        float x2 = q.z + sB1[4 * i + 2];
        float x3 = q.w + sB1[4 * i + 3];
        sum += x0 + x1 + x2 + x3;
        sq += x0 * x0 + x1 * x1 + x2 * x2 + x3 * x3;
      }
      float mean = sum * (1.0f / HID);
      float var = sq * (1.0f / HID) - mean * mean;
      float inv = rsqrtf(var + 1e-5f);
      v8bf* dst = (v8bf*)(sX + lane * HID);
#pragma unroll
      for (int i = 0; i < HID / 8; ++i) {
        float4 q0 = *(const float4*)(hr + 8 * i);
        float4 q1 = *(const float4*)(hr + 8 * i + 4);
        float xv[8] = {q0.x, q0.y, q0.z, q0.w, q1.x, q1.y, q1.z, q1.w};
        v8bf t;
#pragma unroll
        for (int j = 0; j < 8; ++j) {
          int c = 8 * i + j;
          float x = (xv[j] + sB1[c] - mean) * inv * sG1[c] + sBe1[c];
          t[j] = (__bf16)gelu_exact(x);
        }
        dst[i] = t;
      }
    }
    __syncthreads();

    // ---- Phase D: GEMM2  H[32x64] @ W2[64x128] ----
    {
      v8f acc[2][8];
#pragma unroll
      for (int i = 0; i < 2; ++i)
#pragma unroll
        for (int j = 0; j < 8; ++j)
          acc[i][j] = (v8f){0.f, 0.f, 0.f, 0.f, 0.f, 0.f, 0.f, 0.f};
#pragma unroll
      for (int kt = 0; kt < 2; ++kt) {
        v16bf a0 = load_a_frag(sX, HID, 0, kt * 32);
        v16bf a1 = load_a_frag(sX, HID, 16, kt * 32);
#pragma unroll
        for (int nt = 0; nt < 8; ++nt) {
          v16bf bf = load_b_tile(sW2t, kt * 8 + nt);
          acc[0][nt] = wmma_bf16(a0, bf, acc[0][nt]);
          acc[1][nt] = wmma_bf16(a1, bf, acc[1][nt]);
        }
      }
#pragma unroll
      for (int mt = 0; mt < 2; ++mt)
#pragma unroll
        for (int nt = 0; nt < 8; ++nt)
#pragma unroll
          for (int r = 0; r < 8; ++r)
            sY[(mt * 16 + r + half * 8) * COUT + nt * 16 + mr] = acc[mt][nt][r];
    }
    __syncthreads();

    // ---- Phase E: +b2, LayerNorm per row (two-pass, register-light) ----
    {
      float* yr = sY + lane * COUT;
      float sum = 0.f, sq = 0.f;
#pragma unroll
      for (int i = 0; i < COUT / 4; ++i) {
        float4 q = *(const float4*)(yr + 4 * i);
        float x0 = q.x + sB2[4 * i + 0];
        float x1 = q.y + sB2[4 * i + 1];
        float x2 = q.z + sB2[4 * i + 2];
        float x3 = q.w + sB2[4 * i + 3];
        sum += x0 + x1 + x2 + x3;
        sq += x0 * x0 + x1 * x1 + x2 * x2 + x3 * x3;
      }
      float mean = sum * (1.0f / COUT);
      float var = sq * (1.0f / COUT) - mean * mean;
      float inv = rsqrtf(var + 1e-5f);
#pragma unroll
      for (int i = 0; i < COUT / 4; ++i) {
        float4 q = *(const float4*)(yr + 4 * i);
        float4 o;
        o.x = (q.x + sB2[4 * i + 0] - mean) * inv * sG2[4 * i + 0] + sBe2[4 * i + 0];
        o.y = (q.y + sB2[4 * i + 1] - mean) * inv * sG2[4 * i + 1] + sBe2[4 * i + 1];
        o.z = (q.z + sB2[4 * i + 2] - mean) * inv * sG2[4 * i + 2] + sBe2[4 * i + 2];
        o.w = (q.w + sB2[4 * i + 3] - mean) * inv * sG2[4 * i + 3] + sBe2[4 * i + 3];
        *(float4*)(yr + 4 * i) = o;
      }
    }
    __syncthreads();

    // ---- Phase F: K-max + residual LN + GELU; 4 consecutive cols/lane ----
    {
      const int c0 = lane * 4;
      float4 m4 = {-3.4e38f, -3.4e38f, -3.4e38f, -3.4e38f};
#pragma unroll 8
      for (int r = 0; r < KNB; ++r) {
        float4 q = *(const float4*)(sY + r * COUT + c0);
        m4.x = fmaxf(m4.x, q.x); m4.y = fmaxf(m4.y, q.y);
        m4.z = fmaxf(m4.z, q.z); m4.w = fmaxf(m4.w, q.w);
      }
      const int sid = sidx[bm];
      const float* I = feat + ((size_t)b * NPTS + sid) * CIN;
      float4 r4 = *(const float4*)(bres + c0);
#pragma unroll 8
      for (int k = 0; k < CIN; ++k) {
        float fk = I[k];
        float4 w = *(const float4*)(Wres + k * COUT + c0);
        r4.x += fk * w.x; r4.y += fk * w.y; r4.z += fk * w.z; r4.w += fk * w.w;
      }
      float lsum = r4.x + r4.y + r4.z + r4.w;
      float lsq = r4.x * r4.x + r4.y * r4.y + r4.z * r4.z + r4.w * r4.w;
#pragma unroll
      for (int off = 16; off > 0; off >>= 1) {
        lsum += __shfl_xor(lsum, off, 32);
        lsq  += __shfl_xor(lsq, off, 32);
      }
      float rm = lsum * (1.0f / COUT);
      float rvar = lsq * (1.0f / COUT) - rm * rm;
      float rinv = rsqrtf(rvar + 1e-5f);
      float4 g4 = *(const float4*)(gres + c0);
      float4 be4 = *(const float4*)(beres + c0);
      float4 o;
      o.x = gelu_exact(m4.x + (r4.x - rm) * rinv * g4.x + be4.x);
      o.y = gelu_exact(m4.y + (r4.y - rm) * rinv * g4.y + be4.y);
      o.z = gelu_exact(m4.z + (r4.z - rm) * rinv * g4.z + be4.z);
      o.w = gelu_exact(m4.w + (r4.w - rm) * rinv * g4.w + be4.w);
      *(float4*)(outfeat + (size_t)bm * COUT + c0) = o;
    }
    __syncthreads();   // before sX/sH/sY reuse next iteration
  }
}

// ---------------- host launcher --------------------------------------
extern "C" void kernel_launch(void* const* d_in, const int* in_sizes, int n_in,
                              void* d_out, int out_size, void* d_ws, size_t ws_size,
                              hipStream_t stream) {
  const float* pos   = (const float*)d_in[0];
  const float* feat  = (const float*)d_in[1];
  const float* W1    = (const float*)d_in[2];
  const float* b1    = (const float*)d_in[3];
  const float* g1    = (const float*)d_in[4];
  const float* be1   = (const float*)d_in[5];
  const float* W2    = (const float*)d_in[6];
  const float* b2    = (const float*)d_in[7];
  const float* g2    = (const float*)d_in[8];
  const float* be2   = (const float*)d_in[9];
  const float* Wpe   = (const float*)d_in[10];
  const float* bpe   = (const float*)d_in[11];
  const float* Wres  = (const float*)d_in[12];
  const float* bres  = (const float*)d_in[13];
  const float* gres  = (const float*)d_in[14];
  const float* beres = (const float*)d_in[15];

  float* newpos  = (float*)d_out;                 // [B,M,3]
  float* outfeat = newpos + (size_t)BB * MC * 3;  // [B,M,COUT]

  int* sidx = (int*)d_ws;                         // [B*M]
  int* nidx = sidx + BB * MC;                     // [B*M*K]

  fps_kernel<<<BB, 1024, 0, stream>>>(pos, sidx, newpos);
  ballq_kernel<<<(BB * MC * 32) / 256, 256, 0, stream>>>(pos, newpos, nidx);
  mlp_kernel<<<4096, 32, 0, stream>>>(pos, feat, W1, b1, g1, be1, W2, b2, g2, be2,
                                      Wpe, bpe, Wres, bres, gres, beres,
                                      sidx, nidx, newpos, outfeat);
}
